// SparseLogicMachine_90580860273065
// MI455X (gfx1250) — matrix operand order repaired and checked by reading, not compiled, not run on hardware
//
#include <hip/hip_runtime.h>
#include <hip/hip_bf16.h>

typedef _Float16 half_t;
typedef __attribute__((ext_vector_type(16))) _Float16 v16h;
typedef __attribute__((ext_vector_type(2)))  _Float16 h2;
typedef __attribute__((ext_vector_type(8)))  float    v8f;

#define NN 256
#define HIDN 64
#define OUTC 16
#define DINPAD_MAX 128
#define WAVES 4
#define TB (WAVES * 32)

__device__ __forceinline__ float sigmoidf_(float x) {
    return 1.0f / (1.0f + __expf(-x));
}

// A-fragment (16x32, f16) from a row-major f16 tile in LDS.
// Layout per ISA 05_wmma: lanes 0-15 hold M=lane; VGPR v<4 -> K base 0, v>=4 -> K base 16;
// lane-hi half adds +8; each VGPR holds 2 consecutive K halves.
__device__ __forceinline__ v16h load_a_frag(const half_t* F, int ldf, int kb, int lane) {
    const int row = lane & 15;
    const int hi  = (lane >> 4) & 1;
    const half_t* p = F + row * ldf + kb * 32 + hi * 8;
    v16h a;
#pragma unroll
    for (int v = 0; v < 8; ++v) {
        const int kbase = ((v < 4) ? 0 : 16) + (v & 3) * 2;
        h2 t = *(const h2*)(p + kbase);   // 4B-aligned: kbase even, strides even
        a[2 * v]     = t.x;
        a[2 * v + 1] = t.y;
    }
    return a;
}

// Convert fp32 weights [din x ncols] to fragment-swizzled f16.
// dst half index = ((kb*ntc + nt)*32 + lane)*16 + h ;  K = kb*32 + (h<8 ? laneHi*8 + h : 16 + laneHi*8 + h-8) ; col = nt*16 + laneLo
__global__ void convert_w_kernel(const float* __restrict__ src, half_t* __restrict__ dst,
                                 int din, int din_pad, int ntc, int ncols) {
    int tid = blockIdx.x * blockDim.x + threadIdx.x;
    int total = (din_pad >> 5) * ntc * 512;
    if (tid >= total) return;
    int frag = tid >> 9;
    int rem  = tid & 511;
    int lane = rem >> 4;
    int h    = rem & 15;
    int kb   = frag / ntc;
    int nt   = frag % ntc;
    int k    = kb * 32 + ((h < 8) ? ((lane >> 4) & 1) * 8 + h : 16 + ((lane >> 4) & 1) * 8 + (h - 8));
    int col  = nt * 16 + (lane & 15);
    float v  = (k < din && col < ncols) ? src[(size_t)k * ncols + col] : 0.0f;
    dst[tid] = (half_t)v;
}

// Masked exists/forall reduction over axis k of x2[B,N,N,C] -> r2[B,N,2C] (max||min), exclude j==k.
__global__ void reduce2_kernel(const float* __restrict__ x2, float* __restrict__ r2, int C) {
    int g = blockIdx.x * blockDim.x + threadIdx.x;
    int total = 4 * NN * C;
    if (g >= total) return;
    int c = g % C;
    int j = (g / C) % NN;
    int b = g / (C * NN);
    const float* base = x2 + ((size_t)(b * NN + j) * NN) * C + c;
    float mx = -3.4e38f, mn = 3.4e38f;
    for (int k = 0; k < NN; ++k) {
        float v = base[(size_t)k * C];
        mx = fmaxf(mx, (k == j) ? 0.0f : v);
        mn = fminf(mn, (k == j) ? 1.0f : v);
    }
    float* o = r2 + (size_t)(b * NN + j) * (2 * C);
    o[c]     = mx;
    o[C + c] = mn;
}

// Reduction over axis j of x1[B,N,C] -> r1[B,2C]
__global__ void reduce1_kernel(const float* __restrict__ x1, float* __restrict__ r1, int C) {
    int g = blockIdx.x * blockDim.x + threadIdx.x;
    int total = 4 * C;
    if (g >= total) return;
    int c = g % C;
    int b = g / C;
    const float* base = x1 + (size_t)b * NN * C + c;
    float mx = -3.4e38f, mn = 3.4e38f;
    for (int k = 0; k < NN; ++k) {
        float v = base[(size_t)k * C];
        mx = fmaxf(mx, v);
        mn = fminf(mn, v);
    }
    float* o = r1 + b * 2 * C;
    o[c]     = mx;
    o[C + c] = mn;
}

// Fused gather + (logic MLP) * (alpha MLP) per 16-row tile, one tile per wave32.
// mode 2: rows (b,j,k); features [x1[b,j] | x2[b,j,k] | x1[b,k] | x2[b,k,j]]
// mode 1: rows (b,j);   features [x0[b] | x1[b,j] | r2[b,j]]
// mode 0: rows (b);     features [x0[b] | r1[b]]
__global__ __launch_bounds__(TB) void mlp_fused_kernel(
    int mode, int M,
    int C0, int C1, int C2,
    int din, int din_pad,
    const float* __restrict__ x0, const float* __restrict__ x1, const float* __restrict__ x2,
    const float* __restrict__ r1, const float* __restrict__ r2,
    const half_t* __restrict__ w1l, const float* __restrict__ b1l,
    const half_t* __restrict__ w2l, const float* __restrict__ b2l,
    const half_t* __restrict__ w1a, const float* __restrict__ b1a,
    const float* __restrict__ w2a, const float* __restrict__ b2a,
    float* __restrict__ out)
{
    __shared__ __align__(16) half_t sF[WAVES * 16 * DINPAD_MAX];   // 16 KB
    __shared__ __align__(32) half_t sW1[DINPAD_MAX * HIDN];        // 16 KB (one branch at a time)
    __shared__ __align__(16) half_t sH[WAVES * 16 * HIDN];         //  8 KB
    __shared__ __align__(32) half_t sW2[2 * 512];                  //  2 KB (swizzled logic head)
    __shared__ float sW2a[HIDN];
    __shared__ float sB1[HIDN];
    __shared__ float sB2[OUTC];
    __shared__ float sAlpha[WAVES * 16];

    const int tid  = threadIdx.x;
    const int wave = tid >> 5;
    const int lane = tid & 31;
    const int hi   = (lane >> 4) & 1;
    const int row0 = (blockIdx.x * WAVES + wave) * 16;

    if (tid < HIDN) sW2a[tid] = w2a[tid];
    if (tid < OUTC) sB2[tid]  = b2l[tid];
    if (tid < 128)  ((uint4*)sW2)[tid] = ((const uint4*)w2l)[tid];

    // ---- gather this wave's 16 x din_pad feature tile into LDS (f16) ----
    half_t* F = sF + wave * 16 * din_pad;
    for (int r = 0; r < 16; ++r) {
        const int row = row0 + r;
        const bool valid = row < M;
        int b = 0, j = 0, k = 0;
        if (valid) {
            if (mode == 2)      { b = row / (NN * NN); int t = row % (NN * NN); j = t / NN; k = t % NN; }
            else if (mode == 1) { b = row / NN; j = row % NN; }
            else                { b = row; }
        }
        for (int c = lane; c < din_pad; c += 32) {
            float v = 0.0f;
            if (valid && c < din) {
                if (mode == 2) {
                    if (c < C1)              v = x1[(size_t)(b * NN + j) * C1 + c];
                    else if (c < C1 + C2)    v = x2[((size_t)(b * NN + j) * NN + k) * C2 + (c - C1)];
                    else if (c < 2*C1 + C2)  v = x1[(size_t)(b * NN + k) * C1 + (c - C1 - C2)];
                    else                     v = x2[((size_t)(b * NN + k) * NN + j) * C2 + (c - 2*C1 - C2)];
                } else if (mode == 1) {
                    if (c < C0)              v = x0[b * C0 + c];
                    else if (c < C0 + C1)    v = x1[(size_t)(b * NN + j) * C1 + (c - C0)];
                    else                     v = r2[(size_t)(b * NN + j) * (2 * C2) + (c - C0 - C1)];
                } else {
                    if (c < C0)              v = x0[b * C0 + c];
                    else                     v = r1[b * (2 * C1) + (c - C0)];
                }
            }
            F[r * din_pad + c] = (half_t)v;
        }
    }

    const int nkb = din_pad >> 5;
    half_t* Hw = sH + wave * 16 * HIDN;
    v8f logicv = {};

    for (int branch = 0; branch < 2; ++branch) {
        __syncthreads();   // previous users of sW1 done
        {   // stage this branch's swizzled W1 (din_pad x 64 halves) into LDS
            const uint4* src = (const uint4*)(branch ? w1a : w1l);
            uint4* dst = (uint4*)sW1;
            const int n16 = (din_pad * HIDN) >> 3;
            for (int i = tid; i < n16; i += TB) dst[i] = src[i];
        }
        if (tid < HIDN) sB1[tid] = (branch ? b1a : b1l)[tid];
        __syncthreads();

        // ---- GEMM1: F(16 x din_pad) @ W1(din_pad x 64), f16 WMMA, f32 accum ----
        v8f acc[4] = {};
        for (int kb = 0; kb < nkb; ++kb) {
            v16h a = load_a_frag(F, din_pad, kb, lane);
            const v16h* wv = (const v16h*)sW1;
            const int fb = kb * 128 + lane;
#pragma unroll
            for (int nt = 0; nt < 4; ++nt) {
                acc[nt] = __builtin_amdgcn_wmma_f32_16x16x32_f16(
                    false, a, false, wv[fb + nt * 32], (short)0, acc[nt], false, false);
            }
        }

        // bias + ReLU -> h tile (f16) in LDS
#pragma unroll
        for (int nt = 0; nt < 4; ++nt) {
            const int ncol = nt * 16 + (lane & 15);
            const float bias = sB1[ncol];
#pragma unroll
            for (int r = 0; r < 8; ++r) {
                const int m = r + hi * 8;
                float h = acc[nt][r] + bias;
                h = h > 0.0f ? h : 0.0f;
                Hw[m * HIDN + ncol] = (half_t)h;
            }
        }
        __syncthreads();

        if (branch == 0) {
            // ---- GEMM2: h(16x64) @ W2l(64x16) via WMMA, sigmoid ----
            v8f o = {};
#pragma unroll
            for (int kb = 0; kb < 2; ++kb) {
                v16h a2 = load_a_frag(Hw, HIDN, kb, lane);
                o = __builtin_amdgcn_wmma_f32_16x16x32_f16(
                    false, a2, false, ((const v16h*)sW2)[kb * 32 + lane], (short)0, o, false, false);
            }
            const int ncol = lane & 15;
#pragma unroll
            for (int r = 0; r < 8; ++r) o[r] = sigmoidf_(o[r] + sB2[ncol]);
            logicv = o;
        } else {
            // ---- alpha head: 64 -> 1, per-lane VALU dot over the LDS h tile ----
            const int myrow = lane & 15;
            float s = b2a[0];
#pragma unroll
            for (int n = 0; n < HIDN; ++n) s += (float)Hw[myrow * HIDN + n] * sW2a[n];
            if (lane < 16) sAlpha[wave * 16 + lane] = sigmoidf_(s);
        }
        __syncthreads();
    }

    // ---- gate and store: out[row][c] = logic * alpha ----
    const int ncol = lane & 15;
#pragma unroll
    for (int r = 0; r < 8; ++r) {
        const int m = r + hi * 8;
        const int row = row0 + m;
        if (row < M) out[(size_t)row * OUTC + ncol] = logicv[r] * sAlpha[wave * 16 + m];
    }
}

extern "C" void kernel_launch(void* const* d_in, const int* in_sizes, int n_in,
                              void* d_out, int out_size, void* d_ws, size_t ws_size,
                              hipStream_t stream)
{
    (void)out_size; (void)ws_size;
    const int B = 4;

    // Locate x0/x1/x2 vs params (handle params-first flattening defensively).
    int xbase = 0, pbase = 3;
    if (n_in >= 4 && in_sizes[0] != B * 8) { pbase = 0; xbase = n_in - 3; }
    const float* X0in = (const float*)d_in[xbase + 0];
    const float* X1in = (const float*)d_in[xbase + 1];
    const float* X2in = (const float*)d_in[xbase + 2];

    const int C0t[3] = {8, 16, 16}, C1t[3] = {32, 16, 16}, C2t[3] = {32, 16, 16};
    int din_tab[3][3], dp_tab[3][3];
    for (int l = 0; l < 3; ++l) {
        din_tab[l][0] = C0t[l] + 2 * C1t[l];
        din_tab[l][1] = C0t[l] + C1t[l] + 2 * C2t[l];
        din_tab[l][2] = 2 * (C1t[l] + C2t[l]);
        for (int o = 0; o < 3; ++o) dp_tab[l][o] = (din_tab[l][o] + 31) & ~31;
    }

    // Per (layer,order): 8 leaves = two tuples (w1,b1,w2,b2). Detect logic-vs-alpha by w2 size.
    struct Br { const float *w1, *b1, *w2, *b2; };
    Br Lg[3][3], Al[3][3];
    for (int l = 0; l < 3; ++l) for (int o = 0; o < 3; ++o) {
        int base = pbase + (l * 3 + o) * 8;
        bool logicFirst = (in_sizes[base + 2] == 64 * 16);
        int lb = logicFirst ? base : base + 4;
        int ab = logicFirst ? base + 4 : base;
        Lg[l][o] = { (const float*)d_in[lb], (const float*)d_in[lb+1], (const float*)d_in[lb+2], (const float*)d_in[lb+3] };
        Al[l][o] = { (const float*)d_in[ab], (const float*)d_in[ab+1], (const float*)d_in[ab+2], (const float*)d_in[ab+3] };
    }

    // Workspace layout
    char* ws = (char*)d_ws;
    size_t off = 0;
    auto take = [&](size_t bytes) -> size_t {
        size_t a = off; off += (bytes + 255) & ~(size_t)255; return a;
    };

    half_t *w1l_sw[3][3], *w1a_sw[3][3], *w2l_sw[3][3];
    for (int l = 0; l < 3; ++l) for (int o = 0; o < 3; ++o) {
        size_t s = (size_t)dp_tab[l][o] * HIDN * sizeof(half_t);
        w1l_sw[l][o] = (half_t*)(ws + take(s));
        w1a_sw[l][o] = (half_t*)(ws + take(s));
        w2l_sw[l][o] = (half_t*)(ws + take(2 * 512 * sizeof(half_t)));
    }
    float* xA0 = (float*)(ws + take((size_t)B * 16 * 4));
    float* xA1 = (float*)(ws + take((size_t)B * NN * 16 * 4));
    float* xA2 = (float*)(ws + take((size_t)B * NN * NN * 16 * 4));
    float* xB0 = (float*)(ws + take((size_t)B * 16 * 4));
    float* xB1 = (float*)(ws + take((size_t)B * NN * 16 * 4));
    float* xB2 = (float*)(ws + take((size_t)B * NN * NN * 16 * 4));
    float* r2p = (float*)(ws + take((size_t)B * NN * 64 * 4));
    float* r1p = (float*)(ws + take((size_t)B * 64 * 4));

    // Convert all weights to swizzled f16 fragments.
    for (int l = 0; l < 3; ++l) for (int o = 0; o < 3; ++o) {
        int din = din_tab[l][o], dp = dp_tab[l][o];
        int t1 = dp * HIDN; // halves in w1
        convert_w_kernel<<<(t1 + 255) / 256, 256, 0, stream>>>(Lg[l][o].w1, w1l_sw[l][o], din, dp, 4, HIDN);
        convert_w_kernel<<<(t1 + 255) / 256, 256, 0, stream>>>(Al[l][o].w1, w1a_sw[l][o], din, dp, 4, HIDN);
        convert_w_kernel<<<(1024 + 255) / 256, 256, 0, stream>>>(Lg[l][o].w2, w2l_sw[l][o], HIDN, HIDN, 1, OUTC);
    }

    float* out0 = (float*)d_out;
    float* out1 = out0 + (size_t)B * OUTC;
    float* out2 = out1 + (size_t)B * NN * OUTC;

    const float *ix0 = X0in, *ix1 = X1in, *ix2 = X2in;
    for (int l = 0; l < 3; ++l) {
        const int C0 = C0t[l], C1 = C1t[l], C2 = C2t[l];
        float *oy0, *oy1, *oy2;
        if (l == 0)      { oy0 = xA0; oy1 = xA1; oy2 = xA2; }
        else if (l == 1) { oy0 = xB0; oy1 = xB1; oy2 = xB2; }
        else             { oy0 = out0; oy1 = out1; oy2 = out2; }

        reduce2_kernel<<<(B * NN * C2 + 127) / 128, 128, 0, stream>>>(ix2, r2p, C2);
        reduce1_kernel<<<(B * C1 + 63) / 64, 64, 0, stream>>>(ix1, r1p, C1);

        {   // order 0
            int M = B, din = din_tab[l][0], dp = dp_tab[l][0];
            mlp_fused_kernel<<<(M + 63) / 64, TB, 0, stream>>>(
                0, M, C0, C1, C2, din, dp, ix0, ix1, ix2, r1p, r2p,
                w1l_sw[l][0], Lg[l][0].b1, w2l_sw[l][0], Lg[l][0].b2,
                w1a_sw[l][0], Al[l][0].b1, Al[l][0].w2, Al[l][0].b2, oy0);
        }
        {   // order 1
            int M = B * NN, din = din_tab[l][1], dp = dp_tab[l][1];
            mlp_fused_kernel<<<(M + 63) / 64, TB, 0, stream>>>(
                1, M, C0, C1, C2, din, dp, ix0, ix1, ix2, r1p, r2p,
                w1l_sw[l][1], Lg[l][1].b1, w2l_sw[l][1], Lg[l][1].b2,
                w1a_sw[l][1], Al[l][1].b1, Al[l][1].w2, Al[l][1].b2, oy1);
        }
        {   // order 2 (dominant: 4096 workgroups, 16384 wave-tiles)
            int M = B * NN * NN, din = din_tab[l][2], dp = dp_tab[l][2];
            mlp_fused_kernel<<<(M + 63) / 64, TB, 0, stream>>>(
                2, M, C0, C1, C2, din, dp, ix0, ix1, ix2, r1p, r2p,
                w1l_sw[l][2], Lg[l][2].b1, w2l_sw[l][2], Lg[l][2].b2,
                w1a_sw[l][2], Al[l][2].b1, Al[l][2].w2, Al[l][2].b2, oy2);
        }
        ix0 = oy0; ix1 = oy1; ix2 = oy2;
    }
}
